// DeformMSMCConv_32220844655456
// MI455X (gfx1250) — matrix operand order repaired and checked by reading, not compiled, not run on hardware
//
#include <hip/hip_runtime.h>

typedef _Float16 v16h __attribute__((ext_vector_type(16)));
typedef _Float16 v8h  __attribute__((ext_vector_type(8)));
typedef float    v8f  __attribute__((ext_vector_type(8)));
typedef unsigned int u32x4 __attribute__((ext_vector_type(4)));
typedef int          i32x8 __attribute__((ext_vector_type(8)));
typedef int          i32x4 __attribute__((ext_vector_type(4)));

#if defined(__has_builtin)
#  if __has_builtin(__builtin_amdgcn_tensor_load_to_lds) && \
      __has_builtin(__builtin_amdgcn_s_wait_tensorcnt)
#    define USE_TDM 1
#  endif
#endif
#ifndef USE_TDM
#  define USE_TDM 0
#endif

#define B_   4
#define C_   96
#define H_   64
#define W_   64
#define HW_  4096      // H_*W_
#define KOFF 864       // C_*9  (im2col depth of the 3x3 offset conv)
#define CPW  288       // 3*C_  (concat channels)
#define O_   192       // pointwise outputs
#define NTOT 16384     // B_*HW_

// ---------------------------------------------------------------------------
// Convert an f32 matrix [Mvalid x Klen] to f16, zero-padding rows up to Mpad.
// ---------------------------------------------------------------------------
__global__ void cvt_pad_f16(const float* __restrict__ src, _Float16* __restrict__ dst,
                            int Mvalid, int Klen, int total) {
    int i = blockIdx.x * blockDim.x + threadIdx.x;
    if (i >= total) return;
    int m = i / Klen;
    dst[i] = (m < Mvalid) ? (_Float16)src[i] : (_Float16)0.0f;
}

// ---------------------------------------------------------------------------
// Offset conv as implicit GEMM on WMMA with LDS-staged input halo.
// Block = one (b,h) output row, 128 threads (4 waves x 16 columns).
// LDS tile: [96 channels][3 rows][66 cols] f16, columns zero-padded (+-1).
// B fragment gathered ONCE per K-step from LDS, reused across all MT m-tiles.
// ---------------------------------------------------------------------------
template<int K2, int MT>
__global__ __launch_bounds__(128)
void offset_conv_wmma(const float* __restrict__ x, const _Float16* __restrict__ A,
                      const float* __restrict__ bias, float* __restrict__ off) {
    __shared__ _Float16 tile[C_ * 3 * 66];   // 38016 B

    const int tid = threadIdx.x;
    const int bh  = blockIdx.x;              // 0..255
    const int b   = bh >> 6;
    const int h   = bh & 63;
    const float* xb = x + (size_t)b * C_ * HW_;

    // ---- cooperative stage: 288 (c,row) pairs, 16 threads per pair --------
    #pragma unroll 1
    for (int pass = 0; pass < 36; ++pass) {
        int pair = pass * 8 + (tid >> 4);
        int t16  = tid & 15;
        int c    = pair / 3;
        int row  = pair - c * 3;
        int y    = h - 1 + row;
        _Float16* dst = &tile[(c * 3 + row) * 66];
        if ((unsigned)y < (unsigned)H_) {
            const float4 v = reinterpret_cast<const float4*>(xb + c * HW_ + y * W_)[t16];
            dst[1 + t16 * 4 + 0] = (_Float16)v.x;
            dst[1 + t16 * 4 + 1] = (_Float16)v.y;
            dst[1 + t16 * 4 + 2] = (_Float16)v.z;
            dst[1 + t16 * 4 + 3] = (_Float16)v.w;
        } else {
            dst[1 + t16 * 4 + 0] = (_Float16)0.0f;
            dst[1 + t16 * 4 + 1] = (_Float16)0.0f;
            dst[1 + t16 * 4 + 2] = (_Float16)0.0f;
            dst[1 + t16 * 4 + 3] = (_Float16)0.0f;
        }
        if (t16 == 0) { dst[0] = (_Float16)0.0f; dst[65] = (_Float16)0.0f; }
    }
    __syncthreads();

    // ---- WMMA GEMM ---------------------------------------------------------
    const int lane = tid & 31;
    const int wave = tid >> 5;
    const int half = lane >> 4;
    const int wcol = wave * 16 + (lane & 15);   // column w in [0,64)
    const int hw   = h * W_ + wcol;

    v8f acc[MT];
    #pragma unroll
    for (int mi = 0; mi < MT; ++mi) acc[mi] = (v8f){};

    #pragma unroll 1
    for (int kb = 0; kb < KOFF; kb += 32) {
        // B fragment: lane holds col N, K = kb + half*16 + {0..15}; LDS gather
        v16h bf;
        const int kk = kb + half * 16;
        #pragma unroll
        for (int i = 0; i < 16; ++i) {
            int t  = kk + i;
            int c  = t / 9;
            int rs = t - c * 9;
            int r  = rs / 3;
            int s  = rs - r * 3;
            bf[i] = tile[(c * 3 + r) * 66 + wcol + s];
        }
        // reuse B across all m-tiles: MT back-to-back independent WMMAs
        #pragma unroll
        for (int mi = 0; mi < MT; ++mi) {
            const v8h* pa = reinterpret_cast<const v8h*>(
                A + (size_t)(mi * 16 + (lane & 15)) * KOFF + kb + half * 16);
            v8h a0 = pa[0];
            v8h a1 = pa[1];       // +8 halves
            v16h a;
            #pragma unroll
            for (int i = 0; i < 8; ++i) { a[i] = a0[i]; a[8 + i] = a1[i]; }
            acc[mi] = __builtin_amdgcn_wmma_f32_16x16x32_f16(false, a, false, bf,
                                                             (short)0, acc[mi], false, false);
        }
    }

    // D layout: VGPR j, lane -> M = mi*16 + j + 8*half, N = (b,hw)
    #pragma unroll
    for (int mi = 0; mi < MT; ++mi) {
        #pragma unroll
        for (int j = 0; j < 8; ++j) {
            int m = mi * 16 + j + half * 8;
            if (m < 2 * K2)
                off[((size_t)b * (2 * K2) + m) * HW_ + hw] = acc[mi][j] + bias[m];
        }
    }
}

// ---------------------------------------------------------------------------
// Deformable bilinear sampling + depthwise + BN + ReLU, LDS-staged plane.
// Block = (b, c, quarter). The full 16KB channel plane is DMA'd into LDS via
// the Tensor Data Mover (one wave issues tensor_load_to_lds, waits on
// TENSORcnt, then the workgroup barrier releases everyone). All bilinear
// gathers are ds loads; clamping keeps every coordinate inside the plane.
// ---------------------------------------------------------------------------
template<int KS>
__global__ __launch_bounds__(256)
void deform_dw_bnrelu(const float* __restrict__ x, const float* __restrict__ off,
                      const float* __restrict__ wdw,
                      const float* __restrict__ g, const float* __restrict__ be,
                      const float* __restrict__ mm, const float* __restrict__ vv,
                      _Float16* __restrict__ xct, int branch) {
    constexpr int K2 = KS * KS;
    constexpr int R  = (KS - 1) / 2;
    __shared__ float plane[HW_];             // 16 KB

    const int tid = threadIdx.x;
    const int blk = blockIdx.x;              // b*C_*4 + c*4 + q
    const int q   = blk & 3;
    const int bc  = blk >> 2;
    const int c   = bc % C_;
    const int b   = bc / C_;

    const float* xp = x + ((size_t)b * C_ + c) * HW_;
    const float* op = off + (size_t)b * (2 * K2 * HW_);
    __builtin_prefetch(op + q * 1024 + tid, 0, 0);    // gfx1250 global_prefetch_b8

#if USE_TDM
    // ---- stage plane via TDM: D# per ISA 8.3-8.6, 1 row x 4096 x 4B -------
    if (tid < 32) {                                   // one wave issues the DMA
        unsigned long long ga  = (unsigned long long)(uintptr_t)xp;
        unsigned int       lds = (unsigned int)(uintptr_t)plane; // low 32b = LDS offset
        u32x4 g0;
        g0[0] = 1u;                                   // count=1 (valid user D#)
        g0[1] = lds;                                  // lds_addr
        g0[2] = (unsigned int)ga;                     // global_addr[31:0]
        g0[3] = ((unsigned int)(ga >> 32) & 0x01FFFFFFu)  // global_addr[56:32]
              | (2u << 30);                           // type=2 ("image")
        i32x8 g1;
        g1[0] = (2 << 16);                            // data_size=2 (4 bytes)
        g1[1] = (int)(4096u << 16);                   // tensor_dim0=4096 (lo16)
        g1[2] = (1 << 16);                            // dim0 hi | tensor_dim1=1
        g1[3] = (int)(4096u << 16);                   // dim1 hi | tile_dim0=4096
        g1[4] = 1;                                    // tile_dim1=1, tile_dim2=0
        g1[5] = 4096;                                 // tensor_dim0_stride lo
        g1[6] = 0;                                    // stride0 hi | stride1 lo
        g1[7] = 0;
        i32x4 g2;
        g2[0] = 1;                                    // tensor_dim2=1
        g2[1] = 1;                                    // tensor_dim3=1
        g2[2] = 0;                                    // tensor_dim2_stride lo
        g2[3] = 0;                                    // stride hi | tile_dim3=0
        i32x4 g3;
        g3[0] = 0;                                    // tensor_dim3_stride lo
        g3[1] = (1 << 16);                            // stride hi | tensor_dim4=1
        g3[2] = 0;                                    // dim4 hi | tile_dim4=0
        g3[3] = 0;
        i32x8 g4;                                     // extended/unused slot
        g4[0] = 0; g4[1] = 0; g4[2] = 0; g4[3] = 0;
        g4[4] = 0; g4[5] = 0; g4[6] = 0; g4[7] = 0;
        __builtin_amdgcn_tensor_load_to_lds(g0, g1, g2, g3, g4, 0);
        __builtin_amdgcn_s_wait_tensorcnt(0);
    }
#else
    // ---- fallback: cooperative float4 staging ------------------------------
    {
        const float4* src = reinterpret_cast<const float4*>(xp);
        float4* dst = reinterpret_cast<float4*>(plane);
        #pragma unroll
        for (int i = 0; i < 4; ++i) dst[tid + i * 256] = src[tid + i * 256];
    }
#endif
    __syncthreads();

    const float inv  = g[c] * rsqrtf(vv[c] + 1e-5f);
    const float beta = be[c] - mm[c] * inv;

    #pragma unroll 1
    for (int it = 0; it < 4; ++it) {
        const int hw = q * 1024 + it * 256 + tid;      // lanes -> consecutive hw
        const int h  = hw >> 6;
        const int w  = hw & 63;
        float acc = 0.0f;
        #pragma unroll 1
        for (int p = 0; p < K2; ++p) {
            float dy = op[(size_t)p * HW_ + hw];
            float dx = op[(size_t)(K2 + p) * HW_ + hw];
            float py = (float)h + (float)(p / KS - R) + dy;
            float px = (float)w + (float)(p % KS - R) + dx;
            py = fminf(fmaxf(py, 0.0f), 63.0f);
            px = fminf(fmaxf(px, 0.0f), 63.0f);
            float y0f = floorf(py), x0f = floorf(px);
            float wy = py - y0f, wx = px - x0f;
            int y0 = (int)y0f, x0 = (int)x0f;
            int y1 = min(y0 + 1, 63), x1 = min(x0 + 1, 63);
            float v00 = plane[y0 * W_ + x0], v01 = plane[y0 * W_ + x1];
            float v10 = plane[y1 * W_ + x0], v11 = plane[y1 * W_ + x1];
            float top = fmaf(wx, v01 - v00, v00);
            float bot = fmaf(wx, v11 - v10, v10);
            float sv  = fmaf(wy, bot - top, top);
            acc = fmaf(sv, wdw[c * K2 + p], acc);       // uniform -> scalar load
        }
        float r = fmaxf(fmaf(acc, inv, beta), 0.0f);
        int   n = b * HW_ + hw;
        xct[(size_t)n * CPW + branch * C_ + c] = (_Float16)r;
    }
}

// ---------------------------------------------------------------------------
// Pointwise 288->192 GEMM on WMMA, fused BN+ReLU, fp32 output.
// ---------------------------------------------------------------------------
__global__ __launch_bounds__(256)
void pointwise_wmma(const _Float16* __restrict__ xct, const _Float16* __restrict__ wpw,
                    const float* __restrict__ g, const float* __restrict__ be,
                    const float* __restrict__ mm, const float* __restrict__ vv,
                    float* __restrict__ out) {
    const int lane  = threadIdx.x & 31;
    const int wave  = threadIdx.x >> 5;
    const int ntile = blockIdx.x * 8 + wave;
    const int mtile = blockIdx.y;
    const int half  = lane >> 4;

    const int n  = ntile * 16 + (lane & 15);
    const int b  = n >> 12;
    const int hw = n & 4095;

    const _Float16* Arow = wpw + (size_t)(mtile * 16 + (lane & 15)) * CPW;
    const _Float16* Brow = xct + (size_t)n * CPW;

    v8f acc = {};
    #pragma unroll
    for (int kb = 0; kb < CPW; kb += 32) {
        const v8h* pa = reinterpret_cast<const v8h*>(Arow + kb + half * 16);
        v8h a0 = pa[0];
        v8h a1 = pa[1];
        v16h a;
        #pragma unroll
        for (int i = 0; i < 8; ++i) { a[i] = a0[i]; a[8 + i] = a1[i]; }

        const v8h* pb = reinterpret_cast<const v8h*>(Brow + kb + half * 16);
        v8h b0 = pb[0];
        v8h b1 = pb[1];
        v16h bf;
        #pragma unroll
        for (int i = 0; i < 8; ++i) { bf[i] = b0[i]; bf[8 + i] = b1[i]; }

        acc = __builtin_amdgcn_wmma_f32_16x16x32_f16(false, a, false, bf,
                                                     (short)0, acc, false, false);
    }

    #pragma unroll
    for (int j = 0; j < 8; ++j) {
        int   o   = mtile * 16 + j + half * 8;
        float inv = g[o] * rsqrtf(vv[o] + 1e-5f);
        float r   = fmaxf(fmaf(acc[j], inv, be[o] - mm[o] * inv), 0.0f);
        out[((size_t)b * O_ + o) * HW_ + hw] = r;
    }
}

// ---------------------------------------------------------------------------
extern "C" void kernel_launch(void* const* d_in, const int* in_sizes, int n_in,
                              void* d_out, int out_size, void* d_ws, size_t ws_size,
                              hipStream_t stream) {
    const float* x      = (const float*)d_in[0];
    const float* w_off3 = (const float*)d_in[1];
    const float* b_off3 = (const float*)d_in[2];
    const float* w_dw3  = (const float*)d_in[3];
    const float* g3  = (const float*)d_in[4];
    const float* be3 = (const float*)d_in[5];
    const float* m3  = (const float*)d_in[6];
    const float* v3  = (const float*)d_in[7];
    const float* w_off5 = (const float*)d_in[8];
    const float* b_off5 = (const float*)d_in[9];
    const float* w_dw5  = (const float*)d_in[10];
    const float* g5  = (const float*)d_in[11];
    const float* be5 = (const float*)d_in[12];
    const float* m5  = (const float*)d_in[13];
    const float* v5  = (const float*)d_in[14];
    const float* w_off7 = (const float*)d_in[15];
    const float* b_off7 = (const float*)d_in[16];
    const float* w_dw7  = (const float*)d_in[17];
    const float* g7  = (const float*)d_in[18];
    const float* be7 = (const float*)d_in[19];
    const float* m7  = (const float*)d_in[20];
    const float* v7  = (const float*)d_in[21];
    const float* w_pw = (const float*)d_in[22];
    const float* gp = (const float*)d_in[23];
    const float* bp = (const float*)d_in[24];
    const float* mp = (const float*)d_in[25];
    const float* vp = (const float*)d_in[26];

    // Workspace carving (256B aligned slices)
    unsigned char* ws = (unsigned char*)d_ws;
    size_t o = 0;
    auto alloc = [&](size_t bytes) -> void* {
        o = (o + 255) & ~(size_t)255;
        void* p = ws + o;
        o += bytes;
        return p;
    };
    _Float16* A3  = (_Float16*)alloc((size_t)32  * KOFF * 2);
    _Float16* A5  = (_Float16*)alloc((size_t)64  * KOFF * 2);
    _Float16* A7  = (_Float16*)alloc((size_t)112 * KOFF * 2);
    _Float16* WP  = (_Float16*)alloc((size_t)O_  * CPW  * 2);
    float*    off3 = (float*)alloc((size_t)B_ * 18 * HW_ * 4);
    float*    off5 = (float*)alloc((size_t)B_ * 50 * HW_ * 4);
    float*    off7 = (float*)alloc((size_t)B_ * 98 * HW_ * 4);
    _Float16* XCT  = (_Float16*)alloc((size_t)NTOT * CPW * 2);

    // 1) weight conversions to padded f16 A-matrices
    {
        int t;
        t = 32 * KOFF;  cvt_pad_f16<<<(t + 255) / 256, 256, 0, stream>>>(w_off3, A3, 18, KOFF, t);
        t = 64 * KOFF;  cvt_pad_f16<<<(t + 255) / 256, 256, 0, stream>>>(w_off5, A5, 50, KOFF, t);
        t = 112 * KOFF; cvt_pad_f16<<<(t + 255) / 256, 256, 0, stream>>>(w_off7, A7, 98, KOFF, t);
        t = O_ * CPW;   cvt_pad_f16<<<(t + 255) / 256, 256, 0, stream>>>(w_pw, WP, O_, CPW, t);
    }

    // 2) offset convs: 256 blocks (one per output row), LDS halo + WMMA
    offset_conv_wmma<9, 2>  <<<B_ * H_, 128, 0, stream>>>(x, A3, b_off3, off3);
    offset_conv_wmma<25, 4> <<<B_ * H_, 128, 0, stream>>>(x, A5, b_off5, off5);
    offset_conv_wmma<49, 7> <<<B_ * H_, 128, 0, stream>>>(x, A7, b_off7, off7);

    // 3) deformable sampling (TDM-staged LDS plane) + depthwise + BN + ReLU
    {
        int nb = B_ * C_ * 4;
        deform_dw_bnrelu<3><<<nb, 256, 0, stream>>>(x, off3, w_dw3, g3, be3, m3, v3, XCT, 0);
        deform_dw_bnrelu<5><<<nb, 256, 0, stream>>>(x, off5, w_dw5, g5, be5, m5, v5, XCT, 1);
        deform_dw_bnrelu<7><<<nb, 256, 0, stream>>>(x, off7, w_dw7, g7, be7, m7, v7, XCT, 2);
    }

    // 4) pointwise GEMM on WMMA + BN + ReLU
    pointwise_wmma<<<dim3(128, 12), 256, 0, stream>>>(XCT, WP, gp, bp, mp, vp, (float*)d_out);
}
